// SoftGroup_11965778887158
// MI455X (gfx1250) — compile-verified, matrix operand order checked by reading, not run on hardware
//
#include <hip/hip_runtime.h>
#include <hip/hip_bf16.h>

typedef __attribute__((ext_vector_type(16))) _Float16     v16h;
typedef __attribute__((ext_vector_type(8)))  float        v8f;
typedef __attribute__((ext_vector_type(4)))  float        vf4;
typedef __attribute__((ext_vector_type(4)))  unsigned int vu4;

#define P_NUM   128
#define C_ALL   19
#define C_USE   18
#define M_ENT   400000
#define N_PTS   65536
#define WPP     (N_PTS / 32)      // 2048 uint32 words per (class, proposal)
#define CP      (C_USE * P_NUM)   // 2304 (class, proposal) pairs

// ---------------------------------------------------------------------------
// Kernel 1: zero the bitmask workspace (L2-resident, 18.9 MB)
// ---------------------------------------------------------------------------
__global__ void __launch_bounds__(256) zero_bits_k(vu4* __restrict__ bits4, int n4) {
    int i = blockIdx.x * 256 + threadIdx.x;
    if (i < n4) bits4[i] = (vu4){0u, 0u, 0u, 0u};
}

// ---------------------------------------------------------------------------
// Kernel 2: softmax over 19 classes per proposal; stash cls_soft and
// scores_pred = clamp(scores,0,1)*cls_soft laid out [c*P + p] for later reads.
// ---------------------------------------------------------------------------
__global__ void __launch_bounds__(128) softmax_k(const float* __restrict__ cls_scores,
                                                 const float* __restrict__ scores,
                                                 float* __restrict__ smax,
                                                 float* __restrict__ spred) {
    int p = threadIdx.x;
    if (p >= P_NUM) return;
    float v[C_ALL];
    float mx = -3.0e38f;
    for (int c = 0; c < C_ALL; ++c) {
        v[c] = cls_scores[p * C_ALL + c];
        mx = fmaxf(mx, v[c]);
    }
    float s = 0.0f;
    for (int c = 0; c < C_ALL; ++c) {
        v[c] = __expf(v[c] - mx);
        s += v[c];
    }
    float inv = 1.0f / s;
    for (int c = 0; c < C_USE; ++c) {
        float sm = v[c] * inv;
        float sc = fminf(fmaxf(scores[p * C_ALL + c], 0.0f), 1.0f);
        smax[c * P_NUM + p]  = sm;
        spred[c * P_NUM + p] = sc * sm;
    }
}

// ---------------------------------------------------------------------------
// Kernel 3: scatter memberships into per-(c,p) bitmasks. atomicOr implements
// the duplicate-resolving max(...,1) of the reference. ~5M atomics, all L2.
// ---------------------------------------------------------------------------
__global__ void __launch_bounds__(256) scatter_k(const float* __restrict__ mask_scores,
                                                 const int*   __restrict__ prop_ids,
                                                 const int*   __restrict__ point_ids,
                                                 unsigned int* __restrict__ bits) {
    int m = blockIdx.x * 256 + threadIdx.x;
    if (m >= M_ENT) return;
    int prop = prop_ids[m];
    int pt   = point_ids[m];
    unsigned int bit = 1u << (pt & 31);
    int w = pt >> 5;
    const float* ms = mask_scores + (size_t)m * C_ALL;
    #pragma unroll
    for (int c = 0; c < C_USE; ++c) {
        if (ms[c] > -0.5f) {
            atomicOr(&bits[((size_t)(c * P_NUM + prop)) * WPP + w], bit);
        }
    }
}

// ---------------------------------------------------------------------------
// Kernel 4: pointnum via WMMA popcount reduction. One wave32 per (c,p).
// Each step: 32 lanes x 16 words -> popcounts as a 16x32 f16 A tile,
// B = ones  =>  D[m][n] = sum_k A[m][k]  (row sums), accumulated in f32 C.
// 4 chained v_wmma_f32_16x16x32_f16 reduce all 2048 words exactly.
// Lane0 then computes keep / cluster_scores / semantic_id.
// ---------------------------------------------------------------------------
__global__ void __launch_bounds__(256) reduce_k(const unsigned int* __restrict__ bits,
                                                const float* __restrict__ smax,
                                                const float* __restrict__ spred,
                                                float* __restrict__ keepf,
                                                float* __restrict__ out) {
    int wave = (blockIdx.x * 256 + threadIdx.x) >> 5;   // exactly CP waves launched
    int lane = threadIdx.x & 31;

    const unsigned int* base = bits + (size_t)wave * WPP;

    v16h ones;
    #pragma unroll
    for (int j = 0; j < 16; ++j) ones[j] = (_Float16)1.0f;

    v8f acc = {};
    #pragma unroll
    for (int iter = 0; iter < 4; ++iter) {
        const vu4* p4 = (const vu4*)(base + iter * 512 + lane * 16);
        v16h a;
        #pragma unroll
        for (int j = 0; j < 4; ++j) {
            vu4 w = p4[j];
            a[4 * j + 0] = (_Float16)(float)__popc(w[0]);
            a[4 * j + 1] = (_Float16)(float)__popc(w[1]);
            a[4 * j + 2] = (_Float16)(float)__popc(w[2]);
            a[4 * j + 3] = (_Float16)(float)__popc(w[3]);
        }
        acc = __builtin_amdgcn_wmma_f32_16x16x32_f16(
            /*neg_a=*/false, a, /*neg_b=*/false, ones,
            /*c_mod=*/(short)0, acc, /*reuse_a=*/false, /*reuse_b=*/false);
    }

    // Per lane: sum of 8 accumulator rows. All columns identical (B = ones):
    // lanes 0..15 hold rows 0..7, lanes 16..31 hold rows 8..15.
    float s = acc[0] + acc[1] + acc[2] + acc[3] + acc[4] + acc[5] + acc[6] + acc[7];
    float total = __shfl(s, 0, 32) + __shfl(s, 16, 32);

    if (lane == 0) {
        int pointnum = (int)(total + 0.5f);
        float sm = smax[wave];
        float sp = spred[wave];
        bool keep = (sm > 0.001f) && (sp > -1.0f) && (pointnum >= 100);
        float kf = keep ? 1.0f : 0.0f;
        keepf[wave] = kf;

        int c = wave / P_NUM;
        size_t DN = (size_t)CP * N_PTS;
        out[DN + wave]              = keep ? sp : 0.0f;   // cluster_scores
        out[DN + CP + wave]         = kf;                 // keep (bool as 0/1)
        out[DN + 2 * CP + wave]     = (float)(c + 1);     // semantic_id
    }
}

// ---------------------------------------------------------------------------
// Kernel 5: expand bitmask -> dense [C,P,N] floats, masked by keep.
// One float4 per thread, fully coalesced; non-temporal stores keep the
// 604 MB output stream from evicting the L2-resident bitmask.
// ---------------------------------------------------------------------------
__global__ void __launch_bounds__(256) expand_k(const unsigned int* __restrict__ bits,
                                                const float* __restrict__ keepf,
                                                vf4* __restrict__ out4) {
    size_t i = (size_t)blockIdx.x * 256 + threadIdx.x;   // float4 index
    int cp = (int)(i >> 14);          // N/4 = 16384 float4s per (c,p)
    int n4 = (int)(i & 16383);
    int n  = n4 << 2;

    unsigned int w = bits[(size_t)cp * WPP + (n >> 5)];
    float kf = keepf[cp];
    unsigned int sh = (unsigned)(n & 31);

    vf4 v;
    v[0] = ((w >> (sh + 0)) & 1u) ? kf : 0.0f;
    v[1] = ((w >> (sh + 1)) & 1u) ? kf : 0.0f;
    v[2] = ((w >> (sh + 2)) & 1u) ? kf : 0.0f;
    v[3] = ((w >> (sh + 3)) & 1u) ? kf : 0.0f;

    __builtin_nontemporal_store(v, &out4[i]);
}

// ---------------------------------------------------------------------------
extern "C" void kernel_launch(void* const* d_in, const int* in_sizes, int n_in,
                              void* d_out, int out_size, void* d_ws, size_t ws_size,
                              hipStream_t stream) {
    const float* cls_scores  = (const float*)d_in[0];
    const float* scores      = (const float*)d_in[1];
    const float* mask_scores = (const float*)d_in[2];
    const int*   prop_ids    = (const int*)d_in[3];
    const int*   point_ids   = (const int*)d_in[4];
    float* out = (float*)d_out;

    // Workspace layout: bitmask | keepf | smax | spred
    unsigned int* bits = (unsigned int*)d_ws;
    size_t bitsWords = (size_t)CP * WPP;                     // 4,718,592 words
    float* keepf = (float*)((char*)d_ws + bitsWords * 4);
    float* smax  = keepf + CP;
    float* spred = smax + CP;

    int n4 = (int)(bitsWords / 4);                           // 1,179,648
    zero_bits_k<<<(n4 + 255) / 256, 256, 0, stream>>>((vu4*)bits, n4);

    softmax_k<<<1, 128, 0, stream>>>(cls_scores, scores, smax, spred);

    scatter_k<<<(M_ENT + 255) / 256, 256, 0, stream>>>(mask_scores, prop_ids,
                                                       point_ids, bits);

    reduce_k<<<CP / 8, 256, 0, stream>>>(bits, smax, spred, keepf, out);

    size_t tot4 = (size_t)CP * (N_PTS / 4);                  // 37,748,736
    expand_k<<<(unsigned)(tot4 / 256), 256, 0, stream>>>(bits, keepf, (vf4*)out);
}
